// GCN_72507637891474
// MI455X (gfx1250) — matrix operand order repaired and checked by reading, not compiled, not run on hardware
//
#include <hip/hip_runtime.h>

// ---------------- problem constants ----------------
#define NNODES   50000
#define NEDGES   1600000
#define NFEAT    512
#define OUTF     512
#define CLUSTERS 64

typedef __attribute__((ext_vector_type(16))) __bf16 v16bf;
typedef __attribute__((ext_vector_type(8)))  __bf16 v8bf;
typedef __attribute__((ext_vector_type(8)))  float  v8f;

// Split fp32 into bf16 hi + bf16 residual (hi+lo ~ fp32 precision for GEMM)
__device__ __forceinline__ __bf16 bf_hi(float f) { return (__bf16)f; }
__device__ __forceinline__ __bf16 bf_lo(float f, __bf16 hi) {
    return (__bf16)(f - (float)hi);
}

// A-matrix (16x32 bf16) per-lane layout, ISA 7.12.2:
//   lane = half*16 + r ; lane holds row M=r
//   element i -> K = half*8 + i   (i<8) ;  K = 16 + half*8 + (i-8)  (i>=8)
__device__ __forceinline__ void load_a_split(const float* __restrict__ A, int lda,
                                             int row, int k0, int half,
                                             v16bf& ahi, v16bf& alo) {
    const float* p = A + (size_t)row * lda + k0 + half * 8;
    float4 s0 = *reinterpret_cast<const float4*>(p);
    float4 s1 = *reinterpret_cast<const float4*>(p + 4);
    float4 s2 = *reinterpret_cast<const float4*>(p + 16);
    float4 s3 = *reinterpret_cast<const float4*>(p + 20);
    float f[16] = { s0.x, s0.y, s0.z, s0.w, s1.x, s1.y, s1.z, s1.w,
                    s2.x, s2.y, s2.z, s2.w, s3.x, s3.y, s3.z, s3.w };
#pragma unroll
    for (int i = 0; i < 16; ++i) {
        __bf16 hi = bf_hi(f[i]);
        __bf16 lo = bf_lo(f[i], hi);
        ahi[i] = hi;
        alo[i] = lo;
    }
}

// Error-compensated bf16 WMMA: (ahi+alo)(bhi+blo) ~= hi*hi + hi*lo + lo*hi
__device__ __forceinline__ v8f wmma3(const v16bf& ahi, const v16bf& alo,
                                     const v16bf& bhi, const v16bf& blo, v8f c) {
    c = __builtin_amdgcn_wmma_f32_16x16x32_bf16(false, ahi, false, bhi, (short)0, c, false, false);
    c = __builtin_amdgcn_wmma_f32_16x16x32_bf16(false, ahi, false, blo, (short)0, c, false, false);
    c = __builtin_amdgcn_wmma_f32_16x16x32_bf16(false, alo, false, bhi, (short)0, c, false, false);
    return c;
}

// -------- kernel 0a: pack weights into WMMA B-register layout ---------------
// element index = ((kt*NT + tn)*32 + lane)*16 + i  holds
// W[(kt*32 + (lane>>4)*8 + (i<8 ? i : i+8)) * ncols + tn*16 + (lane&15)] as bf16.
__global__ void gcn_pack_w(const float* __restrict__ W, int ncols, int total,
                           __bf16* __restrict__ hi, __bf16* __restrict__ lo) {
    int idx = blockIdx.x * blockDim.x + threadIdx.x;
    if (idx >= total) return;
    const int nt   = ncols >> 4;
    const int i    = idx & 15;
    const int lane = (idx >> 4) & 31;
    const int rest = idx >> 9;
    const int tn   = rest % nt;
    const int kt   = rest / nt;
    const int col  = tn * 16 + (lane & 15);
    const int k    = kt * 32 + (lane >> 4) * 8 + (i < 8 ? i : i + 8);
    const float f  = W[(size_t)k * ncols + col];
    __bf16 h = bf_hi(f);
    hi[idx] = h;
    lo[idx] = bf_lo(f, h);
}

// -------- kernel 0b: pack x into WMMA A-register layout ---------------------
// element index = ((tm*16 + kt)*32 + lane)*16 + i  (16 = NFEAT/32 kt-chunks)
// holds x[tm*16 + (lane&15)][kt*32 + (lane>>4)*8 + (i<8 ? i : i+8)] as bf16.
__global__ void gcn_pack_x(const float* __restrict__ x,
                           __bf16* __restrict__ ahi, __bf16* __restrict__ alo) {
    int idx = blockIdx.x * blockDim.x + threadIdx.x;   // 25,600,000 exact
    const int i    = idx & 15;
    const int lane = (idx >> 4) & 31;
    const int kt   = (idx >> 9) & 15;
    const int tm   = idx >> 13;
    const int half = lane >> 4;
    const int r    = lane & 15;
    const int k    = kt * 32 + half * 8 + (i < 8 ? i : i + 8);
    const float f  = x[(size_t)(tm * 16 + r) * NFEAT + k];
    __bf16 h = bf_hi(f);
    ahi[idx] = h;
    alo[idx] = bf_lo(f, h);
}

__device__ __forceinline__ v16bf load_packA(const __bf16* __restrict__ P,
                                            int tm, int kt, int lane) {
    const __bf16* p = P + (((size_t)tm * 16 + kt) * 32 + lane) * 16;
    return *reinterpret_cast<const v16bf*>(p);
}

__device__ __forceinline__ v16bf load_packB(const __bf16* __restrict__ P,
                                            int kt, int nt, int tn, int lane) {
    const __bf16* p = P + (((size_t)kt * nt + tn) * 32 + lane) * 16;
    return *reinterpret_cast<const v16bf*>(p);
}

// -------- kernel 1: support = x @ W_gc + b_gc -------------------------------
// Inner loop is pure b128 loads + WMMA (A and B both pre-packed bf16 planes).
// Epilogue re-splits fp32 acc into bf16 hi/lo support planes (same
// gather bytes as fp32 in the edge-aggregation stage).
__global__ void gcn_support_gemm(const __bf16* __restrict__ Ahi,
                                 const __bf16* __restrict__ Alo,
                                 const __bf16* __restrict__ Whi,
                                 const __bf16* __restrict__ Wlo,
                                 const float* __restrict__ b,
                                 __bf16* __restrict__ supHi,
                                 __bf16* __restrict__ supLo) {
    const int tile = blockIdx.x;
    const int tm   = tile >> 3;          // 3125 row tiles
    const int tng  = tile & 7;           // 8 groups of 4 column tiles
    const int lane = threadIdx.x & 31;
    const int half = lane >> 4;
    const int r    = lane & 15;

    v8f acc[4];
#pragma unroll
    for (int t = 0; t < 4; ++t) {
        const float bias = b[(tng * 4 + t) * 16 + r];
#pragma unroll
        for (int v = 0; v < 8; ++v) acc[t][v] = bias;
    }

    for (int kt = 0; kt < NFEAT / 32; ++kt) {
        v16bf ahi = load_packA(Ahi, tm, kt, lane);
        v16bf alo = load_packA(Alo, tm, kt, lane);
#pragma unroll
        for (int t = 0; t < 4; ++t) {
            const int tn = tng * 4 + t;
            v16bf bhi = load_packB(Whi, kt, OUTF / 16, tn, lane);
            v16bf blo = load_packB(Wlo, kt, OUTF / 16, tn, lane);
            acc[t] = wmma3(ahi, alo, bhi, blo, acc[t]);
        }
    }

    // C/D layout: VGPR v -> row v (lanes 0-15) / row v+8 (lanes 16-31)
#pragma unroll
    for (int t = 0; t < 4; ++t)
#pragma unroll
        for (int v = 0; v < 8; ++v) {
            const int row = tm * 16 + v + 8 * half;
            const size_t o = (size_t)row * OUTF + (tng * 4 + t) * 16 + r;
            const float f = acc[t][v];
            __bf16 hi = bf_hi(f);
            supHi[o] = hi;
            supLo[o] = bf_lo(f, hi);
        }
}

// -------- kernel 2a: zero h -------------------------------------------------
__global__ void gcn_zero(float4* __restrict__ p, int n4) {
    int i = blockIdx.x * blockDim.x + threadIdx.x;
    if (i < n4) p[i] = make_float4(0.f, 0.f, 0.f, 0.f);
}

// -------- kernel 2b: h[dst] += w * (supHi+supLo)[src]  (one wave per edge) --
// bf16 hi/lo planes: same gather bytes as fp32; h (100 MB) is L2-resident
// so the non-returning f32 atomics resolve in L2.
__global__ void gcn_aggregate(const __bf16* __restrict__ supHi,
                              const __bf16* __restrict__ supLo,
                              const int*   __restrict__ esrc,
                              const int*   __restrict__ edst,
                              const float* __restrict__ ew,
                              float* __restrict__ h) {
    const int e    = (blockIdx.x * blockDim.x + threadIdx.x) >> 5;
    const int lane = threadIdx.x & 31;
    if (e >= NEDGES) return;
    const int   s  = esrc[e];
    const int   d  = edst[e];
    const float wt = ew[e];
    const __bf16* ph = supHi + (size_t)s * OUTF + lane * 16;
    const __bf16* pl = supLo + (size_t)s * OUTF + lane * 16;
    v8bf h0 = *reinterpret_cast<const v8bf*>(ph);
    v8bf h1 = *reinterpret_cast<const v8bf*>(ph + 8);
    v8bf l0 = *reinterpret_cast<const v8bf*>(pl);
    v8bf l1 = *reinterpret_cast<const v8bf*>(pl + 8);
    float* hp = h + (size_t)d * OUTF + lane * 16;
#pragma unroll
    for (int i = 0; i < 8; ++i) {
        atomicAdd(hp + i,     ((float)h0[i] + (float)l0[i]) * wt);
        atomicAdd(hp + 8 + i, ((float)h1[i] + (float)l1[i]) * wt);
    }
}

// -------- kernel 3: logits = normalize(h @ W_low + b_low) -------------------
// One wave owns a 16x64 block (4 WMMA tiles); row L2-norm reduced in-register
// with xor-shuffles inside each 16-lane half.
__global__ void gcn_head_norm(const float* __restrict__ h,
                              const __bf16* __restrict__ Whi,
                              const __bf16* __restrict__ Wlo,
                              const float* __restrict__ b,
                              float* __restrict__ logits) {
    const int tm   = blockIdx.x;
    const int lane = threadIdx.x & 31;
    const int half = lane >> 4;
    const int r    = lane & 15;

    v8f acc[4];
#pragma unroll
    for (int t = 0; t < 4; ++t) {
        const float bias = b[t * 16 + r];
#pragma unroll
        for (int v = 0; v < 8; ++v) acc[t][v] = bias;
    }

    for (int k0 = 0; k0 < OUTF; k0 += 32) {
        v16bf ahi, alo;
        load_a_split(h, OUTF, tm * 16 + r, k0, half, ahi, alo);   // A reused 4x
#pragma unroll
        for (int t = 0; t < 4; ++t) {
            v16bf bhi = load_packB(Whi, k0 >> 5, CLUSTERS / 16, t, lane);
            v16bf blo = load_packB(Wlo, k0 >> 5, CLUSTERS / 16, t, lane);
            acc[t] = wmma3(ahi, alo, bhi, blo, acc[t]);
        }
    }

    float rnorm[8];
#pragma unroll
    for (int v = 0; v < 8; ++v) {
        float s = 0.f;
#pragma unroll
        for (int t = 0; t < 4; ++t) s += acc[t][v] * acc[t][v];
        s += __shfl_xor(s, 1, 32);
        s += __shfl_xor(s, 2, 32);
        s += __shfl_xor(s, 4, 32);
        s += __shfl_xor(s, 8, 32);
        rnorm[v] = fmaxf(sqrtf(s), 1e-12f);
    }

#pragma unroll
    for (int v = 0; v < 8; ++v) {
        const int row = tm * 16 + v + 8 * half;
        const float inv = 1.0f / rnorm[v];
#pragma unroll
        for (int t = 0; t < 4; ++t)
            logits[(size_t)row * CLUSTERS + t * 16 + r] = acc[t][v] * inv;
    }
}

extern "C" void kernel_launch(void* const* d_in, const int* in_sizes, int n_in,
                              void* d_out, int out_size, void* d_ws, size_t ws_size,
                              hipStream_t stream) {
    const float* x    = (const float*)d_in[0];
    const int*   esrc = (const int*)  d_in[1];
    const int*   edst = (const int*)  d_in[2];
    const float* ew   = (const float*)d_in[3];
    const float* Wgc  = (const float*)d_in[4];
    const float* bgc  = (const float*)d_in[5];
    const float* Wlow = (const float*)d_in[6];
    const float* blow = (const float*)d_in[7];

    float* out_h      = (float*)d_out;                       // [NNODES, OUTF]
    float* out_logits = out_h + (size_t)NNODES * OUTF;       // [NNODES, CLUSTERS]

    // ---- workspace layout (all offsets 32B-aligned) ----
    __bf16* supHi  = (__bf16*)d_ws;                          // 51,200,000 B
    __bf16* supLo  = supHi  + (size_t)NNODES * OUTF;         // 51,200,000 B
    __bf16* xAhi   = supLo  + (size_t)NNODES * OUTF;         // 51,200,000 B
    __bf16* xAlo   = xAhi   + (size_t)NNODES * NFEAT;        // 51,200,000 B
    __bf16* WgcHi  = xAlo   + (size_t)NNODES * NFEAT;        //    524,288 B
    __bf16* WgcLo  = WgcHi  + (size_t)NFEAT * OUTF;          //    524,288 B
    __bf16* WlowHi = WgcLo  + (size_t)NFEAT * OUTF;          //     65,536 B
    __bf16* WlowLo = WlowHi + (size_t)OUTF * CLUSTERS;       //     65,536 B

    // 0) pack weights + x into WMMA operand layouts (hi + residual planes)
    gcn_pack_w<<<(NFEAT * OUTF + 255) / 256, 256, 0, stream>>>(Wgc, OUTF, NFEAT * OUTF, WgcHi, WgcLo);
    gcn_pack_w<<<(OUTF * CLUSTERS + 255) / 256, 256, 0, stream>>>(Wlow, CLUSTERS, OUTF * CLUSTERS, WlowHi, WlowLo);
    gcn_pack_x<<<(NNODES * NFEAT) / 256, 256, 0, stream>>>(x, xAhi, xAlo);

    // 1) support = x @ W_gc + b_gc   (3125 * 8 strips of 16x64, 1 wave each)
    gcn_support_gemm<<<(NNODES / 16) * (OUTF / 64), 32, 0, stream>>>(
        xAhi, xAlo, WgcHi, WgcLo, bgc, supHi, supLo);

    // 2) h = segment_sum over edges (zero then atomic accumulate, L2-resident)
    const int n4 = NNODES * OUTF / 4;
    gcn_zero<<<(n4 + 255) / 256, 256, 0, stream>>>((float4*)out_h, n4);
    gcn_aggregate<<<NEDGES / 8, 256, 0, stream>>>(supHi, supLo, esrc, edst, ew, out_h);

    // 3) logits = L2-normalize(h @ W_low + b_low)   (1 wave per 16-row block)
    gcn_head_norm<<<NNODES / 16, 32, 0, stream>>>(out_h, WlowHi, WlowLo, blow, out_logits);
}
